// ChebConv_13125420057165
// MI455X (gfx1250) — matrix operand order repaired and checked by reading, not compiled, not run on hardware
//
#include <hip/hip_runtime.h>

#define N_NODES 50000
#define N_EDGES 800000
#define D_IN    128
#define D_OUT   128
#define K_CHEB  3

// d_ws byte offsets (all 32B-aligned)
#define H_OFF    0ull
#define XHI_OFF  76800000ull   // 3*50000*128*4
#define XLO_OFF  89600000ull   // +50000*128*2
#define WHT_OFF  102400000ull  // +50000*128*2
#define WLT_OFF  102498304ull  // +3*128*128*2

typedef __attribute__((ext_vector_type(16))) __bf16 v16bf;
typedef __attribute__((ext_vector_type(8)))  __bf16 v8bf;
typedef __attribute__((ext_vector_type(8)))  float  v8f;

// ---------------------------------------------------------------------------
// Kernel 1: zero the output accumulator (scatter phase uses atomics into it).
// ---------------------------------------------------------------------------
__global__ __launch_bounds__(256) void cheb_zero_out(float4* __restrict__ p, int n4) {
    int i = blockIdx.x * blockDim.x + threadIdx.x;
    if (i < n4) p[i] = make_float4(0.f, 0.f, 0.f, 0.f);
}

// ---------------------------------------------------------------------------
// Kernel 2a: one-time split of x into bf16 hi/lo planes ([N_NODES, D_IN]).
// ---------------------------------------------------------------------------
__global__ __launch_bounds__(256) void cheb_split_x(
    const float* __restrict__ x, __bf16* __restrict__ Xhi, __bf16* __restrict__ Xlo) {
    int i = blockIdx.x * blockDim.x + threadIdx.x;
    if (i < N_NODES * D_IN) {
        float v  = x[i];
        __bf16 h = (__bf16)v;
        Xhi[i] = h;
        Xlo[i] = (__bf16)(v - (float)h);
    }
}

// ---------------------------------------------------------------------------
// Kernel 2b: one-time split + transpose of W: Wht/Wlt[k][n][kk] (bf16).
// Transposed so the WMMA B-operand K-run is contiguous in memory.
// ---------------------------------------------------------------------------
__global__ __launch_bounds__(256) void cheb_split_w(
    const float* __restrict__ W, __bf16* __restrict__ Wht, __bf16* __restrict__ Wlt) {
    int i = blockIdx.x * blockDim.x + threadIdx.x;
    if (i >= K_CHEB * D_IN * D_OUT) return;
    int k  = i / (D_IN * D_OUT);
    int r  = i - k * (D_IN * D_OUT);
    int kk = r / D_OUT;             // row of W (K dim)
    int n  = r - kk * D_OUT;        // col of W (N dim)
    float v  = W[i];
    __bf16 h = (__bf16)v;
    size_t o = ((size_t)k * D_OUT + n) * D_IN + kk;   // transposed
    Wht[o] = h;
    Wlt[o] = (__bf16)(v - (float)h);
}

// ---------------------------------------------------------------------------
// Kernel 3: H[k] = x @ W[k] via split-bf16 WMMA (fp32-grade accuracy).
// grid = (3125 M-tiles, 3 layers), block = 256 threads = 8 wave32.
// x-tile staged into LDS with GLOBAL_LOAD_ASYNC_TO_LDS_B128 (ASYNCcnt).
// ---------------------------------------------------------------------------
__global__ __launch_bounds__(256) void cheb_gemm_wmma(
    const __bf16* __restrict__ Xhi, const __bf16* __restrict__ Xlo,
    const __bf16* __restrict__ Wht, const __bf16* __restrict__ Wlt,
    float* __restrict__ H)
{
    __shared__ __align__(16) __bf16 Ahi[16 * D_IN];   // 4 KB
    __shared__ __align__(16) __bf16 Alo[16 * D_IN];   // 4 KB

    const int m0 = blockIdx.x * 16;       // node-tile base (exact: 3125*16 = 50000)
    const int k  = blockIdx.y;            // Chebyshev order
    const int t  = threadIdx.x;
    const int lane = t & 31;
    const int wv   = t >> 5;              // 0..7 -> 16-column tile
    const int kh   = lane >> 4;           // lane half (WMMA layout selector)
    const int ln16 = lane & 15;

    // --- Async copy of the 16x128 bf16 hi/lo x-tile (4 KB each) into LDS. ---
    // GVS mode: SGPR 64-bit base + per-lane 32-bit byte offset. 256 threads x
    // 16 B = exactly one tile per instruction. Low 32 bits of a flat LDS
    // pointer are the LDS byte offset (ISA aperture mapping).
    {
        unsigned lds_hi = (unsigned)(uintptr_t)(&Ahi[0]) + t * 16;
        unsigned lds_lo = (unsigned)(uintptr_t)(&Alo[0]) + t * 16;
        unsigned goff   = (unsigned)m0 * (D_IN * 2) + t * 16;
        asm volatile(
            "global_load_async_to_lds_b128 %0, %1, %2\n\t"
            "global_load_async_to_lds_b128 %3, %4, %5\n\t"
            "s_wait_asynccnt 0x0"
            :
            : "v"(lds_hi), "v"(goff), "s"(Xhi),
              "v"(lds_lo), "v"(goff), "s"(Xlo)
            : "memory");
    }
    __syncthreads();

    const __bf16* bh_base = Wht + ((size_t)k * D_OUT + (wv * 16 + ln16)) * D_IN;
    const __bf16* bl_base = Wlt + ((size_t)k * D_OUT + (wv * 16 + ln16)) * D_IN;

    v8f acc = {};

    #pragma unroll
    for (int s = 0; s < 4; ++s) {         // K-steps of 32 over D_IN = 128
        const int kbase = s * 32;

        // A operand (16x32 bf16): lane m = lane&15, half = lane>>4
        // elem e -> k = kbase + half*16 + e  => two 16B runs
        const __bf16* ah = &Ahi[ln16 * D_IN + kbase + (kh << 4)];
        const __bf16* al = &Alo[ln16 * D_IN + kbase + (kh << 4)];
        v8bf ah0 = *(const v8bf*)ah;
        v8bf ah1 = *(const v8bf*)(ah + 8);
        v8bf al0 = *(const v8bf*)al;
        v8bf al1 = *(const v8bf*)(al + 8);
        v16bf a_hi = __builtin_shufflevector(ah0, ah1, 0,1,2,3,4,5,6,7,8,9,10,11,12,13,14,15);
        v16bf a_lo = __builtin_shufflevector(al0, al1, 0,1,2,3,4,5,6,7,8,9,10,11,12,13,14,15);

        // B operand (32x16 bf16): lane n = lane&15, elem e -> k = kbase + half*16 + e
        // Transposed W layout makes this one contiguous 32B run per lane.
        v16bf b_hi = *(const v16bf*)(bh_base + kbase + (kh << 4));
        v16bf b_lo = *(const v16bf*)(bl_base + kbase + (kh << 4));

        // fp32-accurate product: hi*hi + hi*lo + lo*hi (lo*lo ~ 2^-16 rel, dropped)
        acc = __builtin_amdgcn_wmma_f32_16x16x32_bf16(false, a_hi, false, b_hi,
                                                      (short)0, acc, false, false);
        acc = __builtin_amdgcn_wmma_f32_16x16x32_bf16(false, a_hi, false, b_lo,
                                                      (short)0, acc, false, false);
        acc = __builtin_amdgcn_wmma_f32_16x16x32_bf16(false, a_lo, false, b_hi,
                                                      (short)0, acc, false, false);
    }

    // C/D layout: VGPR r -> row m = r + (lane>>4)*8, col n = lane&15
    float* Hk = H + ((size_t)k * N_NODES + m0) * D_OUT;
    #pragma unroll
    for (int r = 0; r < 8; ++r) {
        int m = (kh << 3) + r;
        Hk[(size_t)m * D_OUT + wv * 16 + ln16] = acc[r];
    }
}

// ---------------------------------------------------------------------------
// Kernel 4: out[dst] += w * H[k][src]  — one wave32 per (k, edge).
// Each lane handles 4 consecutive floats (float4 gather + 4 f32 atomics).
// out (25.6 MB) is L2-resident on MI455X (192 MB L2) -> atomics run at L2.
// ---------------------------------------------------------------------------
__global__ __launch_bounds__(256) void cheb_scatter(
    const float* __restrict__ H,           // [K_CHEB, N_NODES, D_OUT]
    const int*   __restrict__ edge_index,  // [K_CHEB, 2, N_EDGES]
    const float* __restrict__ edge_weight, // [K_CHEB, N_EDGES]
    float* __restrict__ out)               // [N_NODES, D_OUT]
{
    const int eg = blockIdx.x * 8 + (threadIdx.x >> 5);   // global (k,edge) wave id
    if (eg >= K_CHEB * N_EDGES) return;
    const int k = eg / N_EDGES;
    const int e = eg - k * N_EDGES;
    const int lane = threadIdx.x & 31;

    const int src = edge_index[(size_t)k * 2 * N_EDGES + e];
    const int dst = edge_index[((size_t)k * 2 + 1) * N_EDGES + e];
    const float w = edge_weight[(size_t)k * N_EDGES + e];

    const float4* hs = (const float4*)(H + ((size_t)k * N_NODES + src) * D_OUT);
    float4 v = hs[lane];

    float* o = out + (size_t)dst * D_OUT + lane * 4;
    atomicAdd(o + 0, v.x * w);
    atomicAdd(o + 1, v.y * w);
    atomicAdd(o + 2, v.z * w);
    atomicAdd(o + 3, v.w * w);
}

// ---------------------------------------------------------------------------
extern "C" void kernel_launch(void* const* d_in, const int* in_sizes, int n_in,
                              void* d_out, int out_size, void* d_ws, size_t ws_size,
                              hipStream_t stream) {
    const float* x  = (const float*)d_in[0];
    const float* W  = (const float*)d_in[1];
    const int*   ei = (const int*)d_in[2];
    const float* ew = (const float*)d_in[3];
    float* out = (float*)d_out;

    char* ws = (char*)d_ws;
    float*  H   = (float*)(ws + H_OFF);     // [K, N, D_OUT] fp32, 76.8 MB
    __bf16* Xhi = (__bf16*)(ws + XHI_OFF);  // [N, D_IN] bf16, 12.8 MB
    __bf16* Xlo = (__bf16*)(ws + XLO_OFF);
    __bf16* Wht = (__bf16*)(ws + WHT_OFF);  // [K, D_OUT, D_IN] bf16, 96 KB
    __bf16* Wlt = (__bf16*)(ws + WLT_OFF);

    // 1) zero output
    {
        int n4 = out_size / 4;   // 6.4M floats -> 1.6M float4
        cheb_zero_out<<<(n4 + 255) / 256, 256, 0, stream>>>((float4*)out, n4);
    }

    // 2) one-time bf16 hi/lo splits (x plain, W transposed)
    {
        int nx = N_NODES * D_IN;                 // 6.4M
        cheb_split_x<<<(nx + 255) / 256, 256, 0, stream>>>(x, Xhi, Xlo);
        int nw = K_CHEB * D_IN * D_OUT;          // 49152
        cheb_split_w<<<(nw + 255) / 256, 256, 0, stream>>>(W, Wht, Wlt);
    }

    // 3) H[k] = x @ W[k] via WMMA
    {
        dim3 grid(N_NODES / 16, K_CHEB);   // 3125 x 3, exact
        cheb_gemm_wmma<<<grid, 256, 0, stream>>>(Xhi, Xlo, Wht, Wlt, H);
    }

    // 4) scatter-accumulate over all edges
    {
        int total_waves = K_CHEB * N_EDGES;        // 2.4M
        int blocks = (total_waves + 7) / 8;        // 8 waves per block
        cheb_scatter<<<blocks, 256, 0, stream>>>(H, ei, ew, out);
    }
}
